// WindowAttention_27513560498656
// MI455X (gfx1250) — matrix operand order repaired and checked by reading, not compiled, not run on hardware
//
#include <hip/hip_runtime.h>

// ---------------- CDNA5 WMMA types ----------------
typedef __attribute__((ext_vector_type(16))) __bf16        bf16x16;
typedef __attribute__((ext_vector_type(2)))  __bf16        bf16x2;
typedef __attribute__((ext_vector_type(8)))  float         floatx8;
typedef __attribute__((ext_vector_type(2)))  float         floatx2;
typedef __attribute__((ext_vector_type(8)))  unsigned int  uintx8;

#define WA_DIM   384
#define WA_HEADS 12
#define WA_HD    32
#define WA_N     49                  // tokens per window (7x7)
#define WA_M     64                  // M padded to 4 WMMA tiles
#define WA_XSZ   (WA_N * WA_DIM)     // 18816
#define WA_NQ    (3 * WA_DIM * WA_DIM)    // qkv_w elems
#define WA_NP    (WA_DIM * WA_DIM)        // proj_w elems
#define WA_NB    (WA_HEADS * WA_N * WA_N) // bias table elems

#if defined(__has_builtin)
#if __has_builtin(__builtin_amdgcn_cvt_pk_bf16_f32)
#define WA_HAVE_CVTPK 1
#endif
#endif

// pack two f32 -> packed bf16 (RNE). Use HW cvt builtin if present, else let
// the backend legalize a vector fptrunc (native v_cvt on gfx1250 if available).
static __device__ __forceinline__ unsigned int packbf(float lo, float hi) {
#ifdef WA_HAVE_CVTPK
    auto p = __builtin_amdgcn_cvt_pk_bf16_f32(lo, hi);
    return __builtin_bit_cast(unsigned int, p);
#else
    floatx2 v = {lo, hi};
    bf16x2 r = __builtin_convertvector(v, bf16x2);
    return __builtin_bit_cast(unsigned int, r);
#endif
}
static __device__ __forceinline__ unsigned short f2bf(float f) {
    __bf16 b = (__bf16)f;
    return __builtin_bit_cast(unsigned short, b);
}

static __device__ __forceinline__ floatx8 wmma_bf16(bf16x16 a, bf16x16 b, floatx8 c) {
    // (neg_a, A, neg_b, B, c_mod, C, reuse_a, reuse_b)
    return __builtin_amdgcn_wmma_f32_16x16x32_bf16(false, a, false, b, (short)0, c, false, false);
}

// A-fragment (16x32 bf16) from LDS bf16 matrix, row-major, stride ld (even).
// Order used here (contiguous): lane m = mo+(L&15); half = L>>4; VGPR j: k = ko+16*half+2j
// -> one contiguous 32B chunk (matches B loaders). Unguarded:
// operand matrices are padded to 64 rows.
static __device__ __forceinline__ bf16x16
load_a_lds(const unsigned short* base, int ld, int mo, int ko, int lane) {
    int m = mo + (lane & 15);
    const unsigned short* row = base + m * ld + ko + 16 * (lane >> 4);
    uintx8 u;
#pragma unroll
    for (int j = 0; j < 4; ++j) {
        u[j]     = *(const unsigned int*)(row + 2 * j);
        u[j + 4] = *(const unsigned int*)(row + 8 + 2 * j);
    }
    return __builtin_bit_cast(bf16x16, u);
}

// B-fragment (32x16 bf16): column n of B == row n of `base` (B^T row-major, LDS).
// lane col n = no+(L&15); half = L>>4; k = ko+16*half+2j -> one 32B chunk.
static __device__ __forceinline__ bf16x16
load_bt_lds(const unsigned short* base, int ld, int no, int ko, int lane) {
    int n = no + (lane & 15);
    const unsigned short* row = base + n * ld + ko + 16 * (lane >> 4);
    uintx8 u;
#pragma unroll
    for (int j = 0; j < 8; ++j) u[j] = *(const unsigned int*)(row + 2 * j);
    return __builtin_bit_cast(bf16x16, u);
}

// B-fragment from pre-converted global bf16 weights (B^T rows, row-major).
static __device__ __forceinline__ bf16x16
load_bt_g16(const unsigned short* base, int ld, int no, int ko, int lane) {
    int n = no + (lane & 15);
    const unsigned short* row = base + (long)n * ld + ko + 16 * (lane >> 4);
    uintx8 u;
#pragma unroll
    for (int j = 0; j < 8; ++j) u[j] = *(const unsigned int*)(row + 2 * j);
    return __builtin_bit_cast(bf16x16, u);
}

// Fallback: B-fragment from global fp32 weights with on-the-fly pack.
static __device__ __forceinline__ bf16x16
load_bt_g32(const float* base, int ld, int no, int ko, int lane) {
    int n = no + (lane & 15);
    const float* row = base + (long)n * ld + ko + 16 * (lane >> 4);
    uintx8 u;
#pragma unroll
    for (int j = 0; j < 8; ++j) u[j] = packbf(row[2 * j], row[2 * j + 1]);
    return __builtin_bit_cast(bf16x16, u);
}

// ---------------- prep: bf16 weights + dense bias table in d_ws ----------------
__global__ __launch_bounds__(256)
void wattn_prep(const float* __restrict__ qkv_w, const float* __restrict__ proj_w,
                const float* __restrict__ rpb, const int* __restrict__ rel,
                unsigned short* __restrict__ wq16, unsigned short* __restrict__ wp16,
                float* __restrict__ bias12) {
    int i = blockIdx.x * 256 + threadIdx.x;
    if (i < WA_NQ) wq16[i] = f2bf(qkv_w[i]);
    if (i < WA_NP) wp16[i] = f2bf(proj_w[i]);
    if (i < WA_NB) {
        int h = i / (WA_N * WA_N), p = i % (WA_N * WA_N);
        bias12[i] = rpb[rel[p] * WA_HEADS + h];
    }
}

// ---------------- fused window-attention kernel ----------------
// One workgroup (8 wave32) per window; x/qkv/probs/O staged in LDS (~128 KB).
// All operand matrices padded to 64 rows; pad contents are either zero (xs, ps —
// correctness-critical) or finite garbage (qs/ks/vts/os pad rows, written
// unguarded; they only ever feed results that are masked by zero probs or
// discarded by the guarded ss/global stores).
template <bool PRE>
__global__ __launch_bounds__(256)
void wattn_fused(const float* __restrict__ x,
                 const float* __restrict__ mask,
                 const float* __restrict__ qkv_w,
                 const float* __restrict__ qkv_b,
                 const float* __restrict__ proj_w,
                 const float* __restrict__ proj_b,
                 const float* __restrict__ rpb,
                 const int*   __restrict__ rel,
                 const unsigned short* __restrict__ wq16,
                 const unsigned short* __restrict__ wp16,
                 const float* __restrict__ bias12,
                 float* __restrict__ out,
                 int nW) {
    __shared__ alignas(16) unsigned short xs[WA_M * WA_DIM];  // x bf16, pad rows = 0
    __shared__ alignas(16) unsigned short os[WA_M * WA_DIM];  // attn out bf16
    __shared__ alignas(16) unsigned short qs[WA_M * WA_HD];   // q (scaled) bf16
    __shared__ alignas(16) unsigned short ks[WA_M * WA_HD];   // k bf16
    __shared__ alignas(16) unsigned short vts[WA_HD * WA_M];  // v^T bf16
    __shared__ alignas(16) unsigned short ps[WA_M * WA_M];    // probs bf16, pad = 0
    __shared__ alignas(16) float          ss[WA_N * 50];      // scores fp32

    const int tid  = threadIdx.x;
    const int lane = tid & 31;
    const int wv   = tid >> 5;
    const long blk = blockIdx.x;
    const int  w   = (int)(blk % nW);
    const float scale = 0.17677669529663687f; // 32^-0.5

    // stage x -> LDS bf16 (non-temporal: streamed once); zero critical pads
    const float* xb = x + blk * WA_XSZ;
    for (int i = tid; i < WA_XSZ; i += 256) xs[i] = f2bf(__builtin_nontemporal_load(xb + i));
    for (int i = WA_XSZ + tid; i < WA_M * WA_DIM; i += 256) xs[i] = 0;
    for (int i = tid; i < WA_M * WA_M; i += 256) ps[i] = 0;
    __syncthreads();

    const float* mw = mask + (long)w * (WA_N * WA_N);
    const int tm = wv >> 1, tn = wv & 1;

    // A-fragments of the x tile: fixed per wave, reused across Q/K/V and all heads
    bf16x16 xa[WA_DIM / 32];
#pragma unroll
    for (int kk = 0; kk < WA_DIM / 32; ++kk)
        xa[kk] = load_a_lds(xs, WA_DIM, tm * 16, kk * 32, lane);

    for (int h = 0; h < WA_HEADS; ++h) {
        // ---- QKV for this head: Q/K/V x (4M x 2N) tiles, 3 per wave ----
#pragma unroll
        for (int s = 0; s < 3; ++s) {
            const long wrow = (long)(s * WA_DIM + h * WA_HD) * WA_DIM;
            floatx8 acc = {};
            bf16x16 b = PRE ? load_bt_g16(wq16 + wrow, WA_DIM, tn * 16, 0, lane)
                            : load_bt_g32(qkv_w + wrow, WA_DIM, tn * 16, 0, lane);
#pragma unroll
            for (int kk = 0; kk < WA_DIM / 32; ++kk) {      // pipelined: prefetch b(kk+1)
                bf16x16 bn = b;
                if (kk + 1 < WA_DIM / 32)
                    bn = PRE ? load_bt_g16(wq16 + wrow, WA_DIM, tn * 16, (kk + 1) * 32, lane)
                             : load_bt_g32(qkv_w + wrow, WA_DIM, tn * 16, (kk + 1) * 32, lane);
                acc = wmma_bf16(xa[kk], b, acc);
                b = bn;
            }
            const int n = tn * 16 + (lane & 15);
            const float bias = qkv_b[s * WA_DIM + h * WA_HD + n];
#pragma unroll
            for (int j = 0; j < 8; ++j) {                   // unguarded LDS stores
                int m = tm * 16 + j + 8 * (lane >> 4);
                float v = acc[j] + bias;
                if      (s == 0) qs[m * WA_HD + n] = f2bf(v * scale);
                else if (s == 1) ks[m * WA_HD + n] = f2bf(v);
                else             vts[n * WA_M + m] = f2bf(v);
            }
        }
        __syncthreads();

        // ---- S = q k^T + rel-pos bias + mask  (16 tiles, 2 per wave) ----
        const float* bh = PRE ? (bias12 + h * (WA_N * WA_N)) : nullptr;
#pragma unroll
        for (int t = wv; t < 16; t += 8) {
            const int stm = t >> 2, stn = t & 3;
            bf16x16 a = load_a_lds(qs, WA_HD, stm * 16, 0, lane);
            bf16x16 b = load_bt_lds(ks, WA_HD, stn * 16, 0, lane);
            floatx8 acc = {};
            acc = wmma_bf16(a, b, acc);
            const int n = stn * 16 + (lane & 15);
            if (n < WA_N) {                                 // guarded: mask/bias bounds
#pragma unroll
                for (int j = 0; j < 8; ++j) {
                    int m = stm * 16 + j + 8 * (lane >> 4);
                    if (m < WA_N) {
                        float bv = PRE ? bh[m * WA_N + n]
                                       : rpb[rel[m * WA_N + n] * WA_HEADS + h];
                        ss[m * 50 + n] = acc[j] + bv + mw[m * WA_N + n];
                    }
                }
            }
        }
        __syncthreads();

        // ---- fp32 softmax, one row per thread; probs written as bf16 ----
        if (tid < WA_N) {
            float* row = ss + tid * 50;
            unsigned short* prow = ps + tid * WA_M;
            float mx = row[0];
            for (int j = 1; j < WA_N; ++j) mx = fmaxf(mx, row[j]);
            float sum = 0.0f;
            for (int j = 0; j < WA_N; ++j) { float e = __expf(row[j] - mx); row[j] = e; sum += e; }
            float inv = 1.0f / sum;
            for (int j = 0; j < WA_N; ++j) prow[j] = f2bf(row[j] * inv);
        }
        __syncthreads();

        // ---- O_h = P V  (8 tiles, 1 per wave; K padded 49->64, pad probs = 0) ----
        {
            floatx8 acc = {};
            bf16x16 a0 = load_a_lds(ps, WA_M, tm * 16, 0, lane);
            bf16x16 b0 = load_bt_lds(vts, WA_M, tn * 16, 0, lane);
            bf16x16 a1 = load_a_lds(ps, WA_M, tm * 16, 32, lane);
            bf16x16 b1 = load_bt_lds(vts, WA_M, tn * 16, 32, lane);
            acc = wmma_bf16(a0, b0, acc);
            acc = wmma_bf16(a1, b1, acc);
            const int n = tn * 16 + (lane & 15);
#pragma unroll
            for (int j = 0; j < 8; ++j) {                   // unguarded LDS stores
                int m = tm * 16 + j + 8 * (lane >> 4);
                os[m * WA_DIM + h * WA_HD + n] = f2bf(acc[j]);
            }
        }
        __syncthreads();   // protects qs/ks/vts/ss/ps reuse next head
    }

    // ---- proj: out = O @ Wp^T + b ----
    // wave -> fixed row-block ptm = wv&3; 12 n-tiles each; A-fragments loaded once.
    const int ptm = wv & 3;
    bf16x16 oa[WA_DIM / 32];
#pragma unroll
    for (int kk = 0; kk < WA_DIM / 32; ++kk)
        oa[kk] = load_a_lds(os, WA_DIM, ptm * 16, kk * 32, lane);

    float* ob = out + blk * WA_XSZ;
#pragma unroll 1                                            // keep code size down
    for (int i = 0; i < 12; ++i) {
        const int ptn = 12 * (wv >> 2) + i;
        const long wrow = (long)(ptn * 16) * WA_DIM;
        floatx8 acc = {};
        bf16x16 b = PRE ? load_bt_g16(wp16 + wrow, WA_DIM, 0, 0, lane)
                        : load_bt_g32(proj_w + wrow, WA_DIM, 0, 0, lane);
#pragma unroll
        for (int kk = 0; kk < WA_DIM / 32; ++kk) {          // pipelined
            bf16x16 bn = b;
            if (kk + 1 < WA_DIM / 32)
                bn = PRE ? load_bt_g16(wp16 + wrow, WA_DIM, 0, (kk + 1) * 32, lane)
                         : load_bt_g32(proj_w + wrow, WA_DIM, 0, (kk + 1) * 32, lane);
            acc = wmma_bf16(oa[kk], b, acc);
            b = bn;
        }
        const int c = ptn * 16 + (lane & 15);
        const float pb = proj_b[c];
#pragma unroll
        for (int j = 0; j < 8; ++j) {
            int m = ptm * 16 + j + 8 * (lane >> 4);
            if (m < WA_N) __builtin_nontemporal_store(acc[j] + pb, ob + m * WA_DIM + c);
        }
    }
}

extern "C" void kernel_launch(void* const* d_in, const int* in_sizes, int n_in,
                              void* d_out, int out_size, void* d_ws, size_t ws_size,
                              hipStream_t stream) {
    const float* x      = (const float*)d_in[0];
    const float* mask   = (const float*)d_in[1];
    const float* qkv_w  = (const float*)d_in[2];
    const float* qkv_b  = (const float*)d_in[3];
    const float* proj_w = (const float*)d_in[4];
    const float* proj_b = (const float*)d_in[5];
    const float* rpb    = (const float*)d_in[6];
    const int*   rel    = (const int*)d_in[7];
    float* out = (float*)d_out;

    const int B_ = in_sizes[0] / WA_XSZ;              // 4096
    const int nW = in_sizes[1] / (WA_N * WA_N);       // 64

    const size_t off_wq = 0;
    const size_t off_wp = off_wq + (size_t)WA_NQ * 2;
    const size_t off_bi = off_wp + (size_t)WA_NP * 2;
    const size_t need   = off_bi + (size_t)WA_NB * 4;   // ~1.3 MB

    if (d_ws != nullptr && ws_size >= need) {
        unsigned short* wq16 = (unsigned short*)((char*)d_ws + off_wq);
        unsigned short* wp16 = (unsigned short*)((char*)d_ws + off_wp);
        float*          bi   = (float*)((char*)d_ws + off_bi);
        wattn_prep<<<dim3((WA_NQ + 255) / 256), dim3(256), 0, stream>>>(
            qkv_w, proj_w, rpb, rel, wq16, wp16, bi);
        wattn_fused<true><<<dim3(B_), dim3(256), 0, stream>>>(
            x, mask, qkv_w, qkv_b, proj_w, proj_b, rpb, rel, wq16, wp16, bi, out, nW);
    } else {
        wattn_fused<false><<<dim3(B_), dim3(256), 0, stream>>>(
            x, mask, qkv_w, qkv_b, proj_w, proj_b, rpb, rel,
            nullptr, nullptr, nullptr, out, nW);
    }
}
